// TransformerEncoderLayer_29317446762569
// MI455X (gfx1250) — matrix operand order repaired and checked by reading, not compiled, not run on hardware
//
#include <hip/hip_runtime.h>
#include <stdint.h>

#define BB 8
#define LL 16384
#define DD 256
#define HH 8
#define DFF 1024
#define TOPK 1000
#define HD 32
#define NTOK (BB*LL)        // 131072
#define M_ROWS (BB*TOPK)    // 8000
#define KPAD 1024
#define NBH (BB*HH)         // 64

typedef __attribute__((ext_vector_type(16))) __bf16 v16bf;
typedef __attribute__((ext_vector_type(8)))  float  v8f;
typedef __attribute__((ext_vector_type(4)))  unsigned int v4u;
typedef __attribute__((ext_vector_type(4)))  int v4i;
typedef __attribute__((ext_vector_type(8)))  int v8i;
typedef unsigned short ushort_t;

struct U128 { unsigned int x, y, z, w; };

#if defined(__HIP_DEVICE_COMPILE__) && __has_builtin(__builtin_amdgcn_tensor_load_to_lds) && __has_builtin(__builtin_amdgcn_s_wait_tensorcnt)
#define HAVE_TDM 1
#endif

__device__ __forceinline__ unsigned short f2bf(float f) {
  unsigned int u = __float_as_uint(f);
  u += 0x7FFFu + ((u >> 16) & 1u);
  return (unsigned short)(u >> 16);
}

// A fragment 16x32 bf16 (MxK). Lane L: row = L&15, K set = {g*8..g*8+7, 16+g*8..16+g*8+7}, g=L>>4.
__device__ __forceinline__ v16bf load_frag_a(const ushort_t* p, int ld, int lane) {
  int g = lane >> 4, m = lane & 15;
  const ushort_t* row = p + m * ld + g * 8;
  union { v16bf v; U128 q[2]; } u;
  u.q[0] = *(const U128*)(row);
  u.q[1] = *(const U128*)(row + 16);
  return u.v;
}
// B fragment 32x16 bf16 (KxN), stored N-major (BT[n][k]). Lane L: col = L&15, K set = {g*8..g*8+7, 16+g*8..16+g*8+7}, g=L>>4.
__device__ __forceinline__ v16bf load_frag_b(const ushort_t* p, int ld, int lane) {
  int n = lane & 15, kk = (lane >> 4) << 3;
  const ushort_t* row = p + n * ld + kk;
  union { v16bf v; U128 q[2]; } u;
  u.q[0] = *(const U128*)(row);
  u.q[1] = *(const U128*)(row + 16);
  return u.v;
}
__device__ __forceinline__ v8f wmma_bf16(v16bf a, v16bf b, v8f c) {
  return __builtin_amdgcn_wmma_f32_16x16x32_bf16(false, a, false, b, (short)0, c, false, false);
}

// TDM 2D tile load: global (2-byte elements, row stride stride0) -> LDS.
// D# built per CDNA5 ISA (Group0/Group1 bitfields).
__device__ __forceinline__ void tdm_load_2d(unsigned int lds_byte, const void* gptr,
                                            unsigned int tile_d0, unsigned int tile_d1,
                                            unsigned int tensor_d0, unsigned int tensor_d1,
                                            unsigned long stride0_elems) {
#ifdef HAVE_TDM
  unsigned long ga = (unsigned long)gptr;
  v4u g0;
  g0.x = 1u;                                                     // count=1 valid descriptor
  g0.y = lds_byte;                                               // lds_addr (bytes)
  g0.z = (unsigned int)(ga & 0xffffffffu);                       // global_addr[31:0]
  g0.w = (unsigned int)((ga >> 32) & 0x01ffffffu) | (2u << 30);  // global_addr[56:32] | type=2
  v8i g1 = {};
  g1[0] = (int)(1u << 16);                                       // data_size=1 (2 bytes)
  g1[1] = (int)((tensor_d0 & 0xffffu) << 16);                    // tensor_dim0[15:0] @ bits63:48
  g1[2] = (int)(((tensor_d0 >> 16) & 0xffffu) | ((tensor_d1 & 0xffffu) << 16));
  g1[3] = (int)(((tensor_d1 >> 16) & 0xffffu) | ((tile_d0 & 0xffffu) << 16));
  g1[4] = (int)(tile_d1 & 0xffffu);                              // tile_dim1; tile_dim2=0
  g1[5] = (int)(unsigned int)(stride0_elems & 0xffffffffu);      // tensor_dim0_stride[31:0]
  g1[6] = (int)(unsigned int)((stride0_elems >> 32) & 0xffffu);  // stride0[47:32]; stride1=0
  g1[7] = 0;
  v4i z4 = {};
#if __clang_major__ >= 23
  v8i z8 = {};
  __builtin_amdgcn_tensor_load_to_lds(g0, g1, z4, z4, z8, 0);
#else
  __builtin_amdgcn_tensor_load_to_lds(g0, g1, z4, z4, 0);
#endif
#else
  (void)lds_byte; (void)gptr; (void)tile_d0; (void)tile_d1;
  (void)tensor_d0; (void)tensor_d1; (void)stride0_elems;
#endif
}
__device__ __forceinline__ void tdm_wait() {
#ifdef HAVE_TDM
  __builtin_amdgcn_s_wait_tensorcnt(0);
#endif
}

// -------------------- copy queries -> out --------------------
__global__ __launch_bounds__(256) void k_copy4(const float4* __restrict__ src,
                                               float4* __restrict__ dst, int n) {
  int i = blockIdx.x * 256 + threadIdx.x;
  if (i < n) dst[i] = src[i];
}

// -------------------- deterministic per-batch top-K --------------------
__global__ __launch_bounds__(1024) void k_topk(const float* __restrict__ sal,
                                               const float* __restrict__ ele,
                                               int* __restrict__ topk) {
  int b = blockIdx.x, tid = threadIdx.x;
  __shared__ unsigned int hist[2048];
  __shared__ unsigned int selbits[512];
  __shared__ int cand_idx[2048];
  __shared__ float cand_s[2048];
  __shared__ unsigned int ccount;
  __shared__ int s_t, s_r;
  __shared__ unsigned int offs[512];
  for (int i = tid; i < 2048; i += 1024) hist[i] = 0u;
  for (int i = tid; i < 512; i += 1024) selbits[i] = 0u;
  if (tid == 0) ccount = 0u;
  __syncthreads();
  const float* sb = sal + (long)b * LL;
  const float* eb = ele + (long)b * LL;
  for (int tk = tid; tk < LL; tk += 1024) {
    float s = sb[tk] + eb[tk];
    int bk = (int)(s * 1024.0f);
    bk = bk < 0 ? 0 : (bk > 2047 ? 2047 : bk);
    atomicAdd(&hist[bk], 1u);
  }
  __syncthreads();
  if (tid == 0) {
    int acc = 0, t = 0;
    for (int bk = 2047; bk >= 0; bk--) {
      int h = (int)hist[bk];
      if (acc + h >= TOPK) { t = bk; break; }
      acc += h;
    }
    s_t = t; s_r = TOPK - acc;
  }
  __syncthreads();
  int t = s_t, r = s_r;
  for (int tk = tid; tk < LL; tk += 1024) {
    float s = sb[tk] + eb[tk];
    int bk = (int)(s * 1024.0f);
    bk = bk < 0 ? 0 : (bk > 2047 ? 2047 : bk);
    if (bk > t) {
      atomicOr(&selbits[tk >> 5], 1u << (tk & 31));
    } else if (bk == t) {
      unsigned int pos = atomicAdd(&ccount, 1u);
      if (pos < 2048u) { cand_idx[pos] = tk; cand_s[pos] = s; }
    }
  }
  __syncthreads();
  int cc = (int)(ccount < 2048u ? ccount : 2048u);
  for (int j = tid; j < cc; j += 1024) {
    float sj = cand_s[j]; int ij = cand_idx[j];
    int rank = 0;
    for (int q = 0; q < cc; q++) {
      float sq = cand_s[q];
      rank += (sq > sj) || (sq == sj && cand_idx[q] < ij);
    }
    if (rank < r) atomicOr(&selbits[ij >> 5], 1u << (ij & 31));
  }
  __syncthreads();
  if (tid == 0) {
    unsigned int acc2 = 0;
    for (int w = 0; w < 512; w++) { offs[w] = acc2; acc2 += (unsigned)__popc(selbits[w]); }
  }
  __syncthreads();
  if (tid < 512) {
    unsigned int bits = selbits[tid];
    unsigned int o = offs[tid];
    while (bits) {
      int bit = __ffs(bits) - 1;
      bits &= bits - 1u;
      if (o < TOPK) topk[b * TOPK + o] = b * LL + tid * 32 + bit;
      o++;
    }
  }
}

// -------------------- f32 -> bf16 transpose (weights, N-major for B operand) --------------------
__global__ __launch_bounds__(256) void k_w2bf_t(const float* __restrict__ src,
                                                ushort_t* __restrict__ dst, int R, int C) {
  int i = blockIdx.x * 256 + threadIdx.x;
  if (i >= R * C) return;
  int r = i / C, c = i % C;
  dst[(long)c * R + r] = f2bf(src[i]);
}

// -------------------- gather + LN1 + rope angles --------------------
__global__ __launch_bounds__(256) void k_gather_ln_rope(
    const float* __restrict__ queries, const int* __restrict__ topk,
    const int* __restrict__ qsi, const int* __restrict__ shapes,
    const float* __restrict__ g, const float* __restrict__ bb,
    float* __restrict__ xg, ushort_t* __restrict__ hbf,
    float* __restrict__ cosb, float* __restrict__ sinb) {
  int r = threadIdx.x >> 4, l16 = threadIdx.x & 15;
  long row = (long)blockIdx.x * 16 + r;
  int tok = topk[row];
  const float* src = queries + (long)tok * DD + l16 * 16;
  float xv[16];
  float sum = 0.f, sq = 0.f;
  for (int j = 0; j < 16; j++) { float v = src[j]; xv[j] = v; sum += v; sq += v * v; }
  for (int m = 1; m < 16; m <<= 1) { sum += __shfl_xor(sum, m, 32); sq += __shfl_xor(sq, m, 32); }
  float mu = sum * (1.0f / 256.0f);
  float var = sq * (1.0f / 256.0f) - mu * mu;
  float inv = rsqrtf(var + 1e-5f);
  for (int j = 0; j < 16; j++) {
    int c = l16 * 16 + j;
    hbf[row * DD + c] = f2bf((xv[j] - mu) * inv * g[c] + bb[c]);
    xg[row * DD + c] = xv[j];
  }
  int p = l16;
  int y = qsi[1 * NTOK + tok], x = qsi[2 * NTOK + tok], lv = qsi[3 * NTOK + tok];
  float ang;
  if (p < 6) {
    float py = (float)y * ((float)shapes[0] / (float)shapes[lv * 2 + 0]);
    ang = py * powf(100.0f, -(float)p / 6.0f);
  } else if (p < 12) {
    float px = (float)x * ((float)shapes[1] / (float)shapes[lv * 2 + 1]);
    ang = px * powf(100.0f, -(float)(p - 6) / 6.0f);
  } else {
    ang = (float)lv * powf(10.0f, -(float)(p - 12) / 4.0f);
  }
  cosb[row * 16 + p] = cosf(ang);
  sinb[row * 16 + p] = sinf(ang);
}

// -------------------- QKV GEMM: (8000x256) @ (256x768) --------------------
__global__ __launch_bounds__(32) void k_gemm_qkv(const ushort_t* __restrict__ A,
                                                 const ushort_t* __restrict__ BT,
                                                 float* __restrict__ Cf) {
  int lane = threadIdx.x & 31;
  int m0 = blockIdx.x * 16, n0 = blockIdx.y * 16;
  v8f c = {};
  for (int kk = 0; kk < 8; kk++) {
    v16bf a = load_frag_a(A + (long)m0 * DD + kk * 32, DD, lane);
    v16bf b = load_frag_b(BT + (long)n0 * DD + kk * 32, DD, lane);
    c = wmma_bf16(a, b, c);
  }
  int n = lane & 15, gg = lane >> 4;
  for (int e = 0; e < 8; e++) {
    long m = m0 + gg * 8 + e;
    Cf[m * 768 + n0 + n] = c[e];
  }
}

// -------------------- rope apply + pack q/k/v into WMMA-friendly layouts --------------------
__global__ __launch_bounds__(256) void k_rope_pack(const float* __restrict__ qkv,
                                                   const float* __restrict__ cosb,
                                                   const float* __restrict__ sinb,
                                                   ushort_t* __restrict__ qp,
                                                   ushort_t* __restrict__ kp,
                                                   ushort_t* __restrict__ vT) {
  long gid = (long)blockIdx.x * 256 + threadIdx.x;  // over B*KPAD*256
  int c = (int)(gid & 255);
  long t2 = gid >> 8;
  int qq = (int)(t2 & 1023);
  int b = (int)(t2 >> 10);
  int h = c >> 5, d = c & 31, p = d >> 1;
  int bh = b * HH + h;
  float qr = 0.f, kr = 0.f, vv = 0.f;
  if (qq < TOPK) {
    long i = (long)b * TOPK + qq;
    float cs = cosb[i * 16 + p], sn = sinb[i * 16 + p];
    float q1 = qkv[i * 768 + c];
    float k1 = qkv[i * 768 + 256 + c];
    vv = qkv[i * 768 + 512 + c];
    float q2 = qkv[i * 768 + (c ^ 1)];
    float k2 = qkv[i * 768 + 256 + (c ^ 1)];
    if ((d & 1) == 0) { qr = q1 * cs - q2 * sn; kr = k1 * cs - k2 * sn; }
    else              { qr = q2 * sn + q1 * cs; kr = k2 * sn + k1 * cs; }
  }
  qp[((long)bh * KPAD + qq) * HD + d] = f2bf(qr);
  kp[((long)bh * KPAD + qq) * HD + d] = f2bf(kr);
  vT[((long)bh * HD + d) * KPAD + qq] = f2bf(vv);
}

// -------------------- flash attention: 1 wave per (bh, 16-query tile) --------------------
__global__ __launch_bounds__(32) void k_attn(const ushort_t* __restrict__ qp,
                                             const ushort_t* __restrict__ kp,
                                             const ushort_t* __restrict__ vT,
                                             ushort_t* __restrict__ outbf) {
  int lane = threadIdx.x & 31;
  int qt = blockIdx.x, bh = blockIdx.y;
  int b = bh >> 3, h = bh & 7;
  int q0 = qt * 16;
  const ushort_t* qpb = qp + (long)bh * KPAD * HD;
  const ushort_t* kpb = kp + (long)bh * KPAD * HD;
  const ushort_t* vtb = vT + (long)bh * HD * KPAD;
  __shared__ alignas(32) ushort_t plds[16 * 32];
  v16bf aq = load_frag_a(qpb + (long)q0 * HD, HD, lane);
  float mx[8], ls[8];
  v8f o0 = {}, o1 = {};
  for (int e = 0; e < 8; e++) { mx[e] = -1e30f; ls[e] = 0.f; }
  const float rs = 0.17677669529663687f;  // 1/sqrt(32)
  int n = lane & 15, gg = lane >> 4;
  for (int c0 = 0; c0 < KPAD; c0 += 32) {
    v16bf kb0 = load_frag_b(kpb + (long)c0 * HD, HD, lane);
    v16bf kb1 = load_frag_b(kpb + (long)(c0 + 16) * HD, HD, lane);
    v8f z = {};
    v8f s0 = wmma_bf16(aq, kb0, z);
    v8f s1 = wmma_bf16(aq, kb1, z);
    bool ok0 = (c0 + n) < TOPK, ok1 = (c0 + 16 + n) < TOPK;
    float p0[8], p1[8];
    for (int e = 0; e < 8; e++) {
      s0[e] = ok0 ? s0[e] * rs : -1e30f;
      s1[e] = ok1 ? s1[e] * rs : -1e30f;
    }
    for (int e = 0; e < 8; e++) {
      float mrow = fmaxf(s0[e], s1[e]);
      for (int msk = 1; msk < 16; msk <<= 1) mrow = fmaxf(mrow, __shfl_xor(mrow, msk, 32));
      float mnew = fmaxf(mx[e], mrow);
      float scl = __expf(mx[e] - mnew);
      mx[e] = mnew;
      float pa = __expf(s0[e] - mnew), pb = __expf(s1[e] - mnew);
      p0[e] = pa; p1[e] = pb;
      float rsum = pa + pb;
      for (int msk = 1; msk < 16; msk <<= 1) rsum += __shfl_xor(rsum, msk, 32);
      ls[e] = ls[e] * scl + rsum;
      o0[e] *= scl; o1[e] *= scl;
    }
    for (int e = 0; e < 8; e++) {
      int m = gg * 8 + e;
      plds[m * 32 + n] = f2bf(p0[e]);
      plds[m * 32 + 16 + n] = f2bf(p1[e]);
    }
    __syncthreads();
    v16bf ap = load_frag_a(plds, 32, lane);
    v16bf bv0 = load_frag_b(vtb + c0, KPAD, lane);
    v16bf bv1 = load_frag_b(vtb + (long)16 * KPAD + c0, KPAD, lane);
    o0 = wmma_bf16(ap, bv0, o0);
    o1 = wmma_bf16(ap, bv1, o1);
    __syncthreads();
  }
  for (int e = 0; e < 8; e++) {
    int m = gg * 8 + e;
    int q = q0 + m;
    if (q < TOPK) {
      long row = (long)b * TOPK + q;
      float invl = 1.0f / ls[e];
      outbf[row * DD + h * HD + n] = f2bf(o0[e] * invl);
      outbf[row * DD + h * HD + 16 + n] = f2bf(o1[e] * invl);
    }
  }
}

// -------------------- Wo GEMM + residual + scatter into out --------------------
__global__ __launch_bounds__(32) void k_gemm_wo_scatter(const ushort_t* __restrict__ A,
                                                        const ushort_t* __restrict__ WoT,
                                                        const float* __restrict__ xg,
                                                        const int* __restrict__ topk,
                                                        float* __restrict__ outq) {
  int lane = threadIdx.x & 31;
  int m0 = blockIdx.x * 16, n0 = blockIdx.y * 16;
  v8f c = {};
  for (int kk = 0; kk < 8; kk++) {
    v16bf a = load_frag_a(A + (long)m0 * DD + kk * 32, DD, lane);
    v16bf b = load_frag_b(WoT + (long)n0 * DD + kk * 32, DD, lane);
    c = wmma_bf16(a, b, c);
  }
  int n = lane & 15, gg = lane >> 4;
  for (int e = 0; e < 8; e++) {
    int m = m0 + gg * 8 + e;
    int col = n0 + n;
    int tok = topk[m];
    outq[(long)tok * DD + col] = c[e] + xg[(long)m * DD + col];
  }
}

// -------------------- fused LN2 + FFN, 32 rows/block, TDM-staged W1 chunks --------------------
#define FCH 64  // DFF chunk per iteration
__global__ __launch_bounds__(256) void k_ffn(float* __restrict__ outq,
                                             const float* __restrict__ lng,
                                             const float* __restrict__ lnb,
                                             const ushort_t* __restrict__ W1T,
                                             const float* __restrict__ b1,
                                             const ushort_t* __restrict__ W2T,
                                             const float* __restrict__ b2) {
  __shared__ alignas(32) ushort_t h2s[32 * 256];   // 16KB LN'd activations (bf16)
  __shared__ alignas(32) ushort_t acts[32 * FCH];  // 4KB  gelu activations (bf16)
  __shared__ alignas(32) ushort_t w1s[FCH * 256];  // 32KB W1 chunk staged by TDM
  int t = threadIdx.x;
  long row0 = (long)blockIdx.x * 32;
  // ---- LN2: 32 rows, 8 threads per row ----
  {
    int r = t >> 3, l8 = t & 7;
    const float* src = outq + (row0 + r) * DD + l8 * 32;
    float xv[32];
    float sum = 0.f, sq = 0.f;
    for (int j = 0; j < 32; j++) { float v = src[j]; xv[j] = v; sum += v; sq += v * v; }
    for (int m = 1; m < 8; m <<= 1) { sum += __shfl_xor(sum, m, 32); sq += __shfl_xor(sq, m, 32); }
    float mu = sum * (1.0f / 256.0f);
    float var = sq * (1.0f / 256.0f) - mu * mu;
    float inv = rsqrtf(var + 1e-5f);
    for (int j = 0; j < 32; j++) {
      int c = l8 * 32 + j;
      h2s[r * 256 + c] = f2bf((xv[j] - mu) * inv * lng[c] + lnb[c]);
    }
  }
  int lane = t & 31, w = t >> 5;  // 8 waves
  int n = lane & 15, gg = lane >> 4;
  v8f acc[4] = {{}, {}, {}, {}};  // [mt*2 + j], output cols (2w+j)*16, rows mt*16
  unsigned int w1s_lds = (unsigned int)(unsigned long)(const void*)&w1s[0];
  for (int it = 0; it < DFF / FCH; it++) {
    int dff0 = it * FCH;
    __syncthreads();  // prior iter done reading w1s/acts (also covers LN2->h2s first time)
#ifdef HAVE_TDM
    if (w == 0) {
      tdm_load_2d(w1s_lds, W1T + (long)dff0 * DD, /*tile_d0=*/DD, /*tile_d1=*/FCH,
                  /*tensor_d0=*/DD, /*tensor_d1=*/FCH, /*stride0=*/DD);
      tdm_wait();
    }
#else
    for (int i = t; i < FCH * DD / 8; i += 256)
      ((U128*)w1s)[i] = ((const U128*)(W1T + (long)dff0 * DD))[i];
#endif
    __syncthreads();  // w1s ready
    // ---- stage 1: G = h2(32x256) @ W1chunk(256x64); 8 tiles -> 1 per wave ----
    {
      int mt = w >> 2, nt = w & 3;
      v8f gf = {};
      for (int kk = 0; kk < 8; kk++) {
        v16bf a = load_frag_a((const ushort_t*)h2s + mt * 16 * 256 + kk * 32, 256, lane);
        v16bf b = load_frag_b((const ushort_t*)w1s + (nt * 16) * 256 + kk * 32, 256, lane);
        gf = wmma_bf16(a, b, gf);
      }
      for (int e = 0; e < 8; e++) {
        int m = mt * 16 + gg * 8 + e;
        int col = dff0 + nt * 16 + n;
        float xgv = gf[e] + b1[col];
        float x3 = xgv * xgv * xgv;
        float gl = 0.5f * xgv * (1.0f + tanhf(0.7978845608028654f * (xgv + 0.044715f * x3)));
        acts[m * FCH + nt * 16 + n] = f2bf(gl);
      }
    }
    __syncthreads();  // acts ready
    // ---- stage 2: acc += acts(32x64) @ W2chunk(64x256); wave w: n-tiles 2w,2w+1 x m-tiles 0,1
    for (int kk = 0; kk < 2; kk++) {
      v16bf a0 = load_frag_a((const ushort_t*)acts + 0 * 16 * FCH + kk * 32, FCH, lane);
      v16bf a1 = load_frag_a((const ushort_t*)acts + 1 * 16 * FCH + kk * 32, FCH, lane);
      for (int j = 0; j < 2; j++) {
        v16bf b = load_frag_b(W2T + (long)((2 * w + j) * 16) * DFF + dff0 + kk * 32, DFF, lane);
        acc[0 * 2 + j] = wmma_bf16(a0, b, acc[0 * 2 + j]);
        acc[1 * 2 + j] = wmma_bf16(a1, b, acc[1 * 2 + j]);
      }
    }
  }
  // ---- epilogue: residual + bias ----
  for (int mt = 0; mt < 2; mt++) {
    for (int j = 0; j < 2; j++) {
      for (int e = 0; e < 8; e++) {
        long rr = row0 + mt * 16 + gg * 8 + e;
        int col = (2 * w + j) * 16 + n;
        outq[rr * DD + col] = outq[rr * DD + col] + acc[mt * 2 + j][e] + b2[col];
      }
    }
  }
}

extern "C" void kernel_launch(void* const* d_in, const int* in_sizes, int n_in,
                              void* d_out, int out_size, void* d_ws, size_t ws_size,
                              hipStream_t stream) {
  (void)in_sizes; (void)n_in; (void)out_size; (void)ws_size;
  const float* queries = (const float*)d_in[0];
  const float* sal     = (const float*)d_in[2];
  const float* ele     = (const float*)d_in[3];
  const int*   qsi     = (const int*)d_in[4];
  const int*   shapes  = (const int*)d_in[6];
  const float* Wqkv    = (const float*)d_in[7];
  const float* Wo      = (const float*)d_in[8];
  const float* ln1g    = (const float*)d_in[9];
  const float* ln1b    = (const float*)d_in[10];
  const float* W1      = (const float*)d_in[11];
  const float* b1      = (const float*)d_in[12];
  const float* W2      = (const float*)d_in[13];
  const float* b2      = (const float*)d_in[14];
  const float* ln2g    = (const float*)d_in[15];
  const float* ln2b    = (const float*)d_in[16];
  float* outq = (float*)d_out;

  char* ws = (char*)d_ws;
  size_t off = 0;
  auto alloc = [&](size_t bytes) -> char* {
    char* p = ws + off; off = (off + bytes + 255) & ~(size_t)255; return p;
  };
  int*      topk  = (int*)alloc((size_t)M_ROWS * 4);
  float*    xg    = (float*)alloc((size_t)M_ROWS * DD * 4);
  ushort_t* hbf   = (ushort_t*)alloc((size_t)M_ROWS * DD * 2);
  float*    cosb  = (float*)alloc((size_t)M_ROWS * 16 * 4);
  float*    sinb  = (float*)alloc((size_t)M_ROWS * 16 * 4);
  ushort_t* WqkvT = (ushort_t*)alloc((size_t)768 * DD * 2);
  ushort_t* WoT   = (ushort_t*)alloc((size_t)DD * DD * 2);
  ushort_t* W1T   = (ushort_t*)alloc((size_t)DFF * DD * 2);
  ushort_t* W2T   = (ushort_t*)alloc((size_t)DD * DFF * 2);
  float*    qkvf  = (float*)alloc((size_t)M_ROWS * 768 * 4);
  ushort_t* qp    = (ushort_t*)alloc((size_t)NBH * KPAD * HD * 2);
  ushort_t* kp    = (ushort_t*)alloc((size_t)NBH * KPAD * HD * 2);
  ushort_t* vT    = (ushort_t*)alloc((size_t)NBH * HD * KPAD * 2);
  ushort_t* outbf = (ushort_t*)alloc((size_t)M_ROWS * DD * 2);

  {
    int n4 = NTOK * DD / 4;
    k_copy4<<<dim3((n4 + 255) / 256), dim3(256), 0, stream>>>((const float4*)queries, (float4*)outq, n4);
  }
  k_topk<<<dim3(BB), dim3(1024), 0, stream>>>(sal, ele, topk);
  k_w2bf_t<<<dim3((DD * 768 + 255) / 256), dim3(256), 0, stream>>>(Wqkv, WqkvT, DD, 768);
  k_w2bf_t<<<dim3((DD * DD + 255) / 256), dim3(256), 0, stream>>>(Wo, WoT, DD, DD);
  k_w2bf_t<<<dim3((DD * DFF + 255) / 256), dim3(256), 0, stream>>>(W1, W1T, DD, DFF);
  k_w2bf_t<<<dim3((DFF * DD + 255) / 256), dim3(256), 0, stream>>>(W2, W2T, DFF, DD);
  k_gather_ln_rope<<<dim3(M_ROWS / 16), dim3(256), 0, stream>>>(
      queries, topk, qsi, shapes, ln1g, ln1b, xg, hbf, cosb, sinb);
  k_gemm_qkv<<<dim3(M_ROWS / 16, 48), dim3(32), 0, stream>>>(hbf, WqkvT, qkvf);
  {
    int nt = BB * KPAD * 256;
    k_rope_pack<<<dim3(nt / 256), dim3(256), 0, stream>>>(qkvf, cosb, sinb, qp, kp, vT);
  }
  k_attn<<<dim3((TOPK + 15) / 16, NBH), dim3(32), 0, stream>>>(qp, kp, vT, outbf);
  k_gemm_wo_scatter<<<dim3(M_ROWS / 16, DD / 16), dim3(32), 0, stream>>>(outbf, WoT, xg, topk, outq);
  k_ffn<<<dim3(NTOK / 32), dim3(256), 0, stream>>>(outq, ln2g, ln2b, W1T, b1, W2T, b2);
}
